// BinarizeLinear_inference_10402410791022
// MI455X (gfx1250) — matrix operand order repaired and checked by reading, not compiled, not run on hardware
//
#include <hip/hip_runtime.h>

// ---------------------------------------------------------------------------
// BinarizeLinear: out = sign(x) @ W + bias          (8192x4096 @ 4096x4096)
//
// Phase 1a: binarize X -> exact {+1,-1} bf16 in ws           (64 MB)
// Phase 1b: convert+transpose W -> bf16 Wt[n][k] in ws       (32 MB)
// Phase 2 : bf16 WMMA GEMM; A/B tiles DMA'd into LDS by the Tensor Data
//           Mover (TENSOR_LOAD_TO_LDS, TENSORcnt), double-buffered so DMA
//           overlaps v_wmma_f32_16x16x32_bf16 compute.
// ---------------------------------------------------------------------------

typedef __attribute__((ext_vector_type(16))) __bf16        v16bf;
typedef __attribute__((ext_vector_type(8)))  float         v8f;
typedef __attribute__((ext_vector_type(4)))  unsigned int  v4u;
typedef __attribute__((ext_vector_type(8)))  int           v8i;
typedef __attribute__((ext_vector_type(4)))  int           v4i;

#define TOKENS 8192
#define IN_F   4096
#define OUT_F  4096

#define BM 128
#define BN 128
#define BK 64
// TDM pads 4 dwords after every 32 dwords (=64 bf16 = one K-row of a tile),
// so each tile row/col lands at a 144-byte stride in LDS: 16B-aligned and
// bank-conflict-free for the b128 fragment reads below.
#define LDS_STRIDE 72                  /* ushorts = 144 B                   */
#define TILE_USHORTS (128 * LDS_STRIDE) /* 9216 ushorts = 18 KB per tile   */

__device__ __forceinline__ unsigned short f2bf_bits(float f) {
  return __builtin_bit_cast(unsigned short, (__bf16)f);
}
__device__ __forceinline__ unsigned int sign_bf16(float f) {
  return (f > 0.0f) ? 0x3F80u : 0xBF80u;  // exact +-1.0 in bf16
}
__device__ __forceinline__ unsigned lds_offset(const void* p) {
  // Generic (flat) LDS address: bits [31:0] are the workgroup LDS byte offset.
  return (unsigned)(unsigned long long)(uintptr_t)p;
}

union Frag16 {
  uint4 q[2];
  v16bf v;
};

// ---------------------------------------------------------------------------
// 2D TDM load: tile_dim0=BK (K, contiguous), tile_dim1=128 (rows), 2B elems,
// row stride = 4096 elements, LDS padding 4 dwords per 32 dwords.
// D# packing per CDNA5 ISA sec 8.3/8.4; groups 2/3 zero (2D tensor).
// ---------------------------------------------------------------------------
__device__ __forceinline__ void tdm_load_tile(unsigned lds_off,
                                              unsigned long long gaddr,
                                              unsigned tensor_rows) {
  v4u g0;
  g0[0] = 1u;                                    // count=1 (valid), load, user
  g0[1] = lds_off;                               // lds_addr
  g0[2] = (unsigned)gaddr;                       // global_addr[31:0]
  g0[3] = (unsigned)(gaddr >> 32) | (2u << 30);  // global_addr[56:32] | type=2

  v8i g1;
  g1[0] = (int)((1u << 16)      // data_size = 1 -> 2 bytes
              | (1u << 20)      // pad_enable
              | (4u << 22)      // pad_interval: 32 dwords (=128B data row)
              | (3u << 25));    // pad_amount: 4 dwords (=16B)
  g1[1] = (int)(((unsigned)IN_F & 0xFFFFu) << 16);        // tensor_dim0 lo16
  g1[2] = (int)(((unsigned)IN_F >> 16) |
                ((tensor_rows & 0xFFFFu) << 16));         // dim0 hi | dim1 lo
  g1[3] = (int)((tensor_rows >> 16) | ((unsigned)BK << 16)); // dim1 hi|tile0
  g1[4] = (int)(128u);                                    // tile_dim1 | tile2=0
  g1[5] = (int)((unsigned)IN_F);                          // dim0_stride lo32
  g1[6] = 0;                                              // stride hi | d1s lo
  g1[7] = 0;

  v4i gz = (v4i){0, 0, 0, 0};
#if __clang_major__ >= 23
  v8i gz8 = (v8i){0, 0, 0, 0, 0, 0, 0, 0};
  __builtin_amdgcn_tensor_load_to_lds(g0, g1, gz, gz, gz8, 0);
#else
  __builtin_amdgcn_tensor_load_to_lds(g0, g1, gz, gz, 0);
#endif
}

// ---------------------------------------------------------------------------
// Phase 1a: binarize X to bf16 {+1,-1}; 8 elements/thread, b128 stores.
// ---------------------------------------------------------------------------
__global__ __launch_bounds__(256) void xbin_kernel(
    const float* __restrict__ X, unsigned short* __restrict__ Xb) {
  const size_t i = ((size_t)blockIdx.x * 256 + threadIdx.x) * 8;
  const float4 a = *(const float4*)(X + i);
  const float4 b = *(const float4*)(X + i + 4);
  uint4 o;
  o.x = sign_bf16(a.x) | (sign_bf16(a.y) << 16);
  o.y = sign_bf16(a.z) | (sign_bf16(a.w) << 16);
  o.z = sign_bf16(b.x) | (sign_bf16(b.y) << 16);
  o.w = sign_bf16(b.z) | (sign_bf16(b.w) << 16);
  *(uint4*)(Xb + i) = o;
}

// ---------------------------------------------------------------------------
// Phase 1b: W[k][n] fp32 -> Wt[n][k] bf16 (32x32 LDS tile transpose).
// ---------------------------------------------------------------------------
__global__ __launch_bounds__(256) void wtrans_kernel(
    const float* __restrict__ W, unsigned short* __restrict__ Wt) {
  __shared__ unsigned short tile[32][33];
  const int k0 = blockIdx.y * 32;
  const int n0 = blockIdx.x * 32;
  const int c  = threadIdx.x & 31;
  const int r0 = threadIdx.x >> 5;  // 0..7
#pragma unroll
  for (int j = 0; j < 32; j += 8) {
    const int r = r0 + j;
    tile[c][r] = f2bf_bits(W[(size_t)(k0 + r) * OUT_F + (n0 + c)]);
  }
  __syncthreads();
#pragma unroll
  for (int j = 0; j < 32; j += 8) {
    const int r = r0 + j;
    Wt[(size_t)(n0 + r) * IN_F + (k0 + c)] = tile[r][c];
  }
}

// ---------------------------------------------------------------------------
// Phase 2: TDM + WMMA GEMM.  128x128 block, BK=64, 8 waves in 2x4 grid,
// wave tile 64x32 -> 16 WMMAs per K-tile.  Double-buffered LDS (72 KB).
// ---------------------------------------------------------------------------
__global__ __launch_bounds__(256) void binlin_tdm_wmma_kernel(
    const unsigned short* __restrict__ Xb, const unsigned short* __restrict__ Wt,
    const float* __restrict__ Bias, float* __restrict__ O) {
  __shared__ __align__(16) unsigned short lds[2][2 * TILE_USHORTS];  // 72 KB

  const int t    = threadIdx.x;
  const int lane = t & 31;
  const int wave = t >> 5;     // 0..7
  const int wm   = wave >> 2;  // 0..1 : 64-row slab
  const int wn   = wave & 3;   // 0..3 : 32-col slab
  const int l16  = lane & 15;
  const int lhi  = lane >> 4;  // 0/1 half-wave

  const int bm = blockIdx.y * BM;
  const int bn = blockIdx.x * BN;

  const unsigned long long gaA0 =
      (unsigned long long)(uintptr_t)Xb + ((size_t)bm * IN_F) * 2ull;
  const unsigned long long gaB0 =
      (unsigned long long)(uintptr_t)Wt + ((size_t)bn * IN_F) * 2ull;
  unsigned ldsA_off[2], ldsB_off[2];
#pragma unroll
  for (int bf = 0; bf < 2; ++bf) {
    ldsA_off[bf] = lds_offset(&lds[bf][0]);
    ldsB_off[bf] = lds_offset(&lds[bf][TILE_USHORTS]);
  }

  v8f acc[4][2];
#pragma unroll
  for (int tm = 0; tm < 4; ++tm)
#pragma unroll
    for (int tn = 0; tn < 2; ++tn)
      acc[tm][tn] = (v8f){};

  const int NT = IN_F / BK;  // 64 K-tiles

  // Prologue: DMA tile 0 into buffer 0.
  if (wave == 0) {
    tdm_load_tile(ldsA_off[0], gaA0, (unsigned)TOKENS);
    tdm_load_tile(ldsB_off[0], gaB0, (unsigned)OUT_F);
    __builtin_amdgcn_s_wait_tensorcnt(0);
  }
  __syncthreads();

  for (int i = 0; i < NT; ++i) {
    const int cur = i & 1;
    // Kick off next tile's DMA into the other buffer (overlaps WMMA below).
    if (wave == 0 && (i + 1) < NT) {
      const unsigned long long koff = (unsigned long long)(i + 1) * BK * 2ull;
      tdm_load_tile(ldsA_off[cur ^ 1], gaA0 + koff, (unsigned)TOKENS);
      tdm_load_tile(ldsB_off[cur ^ 1], gaB0 + koff, (unsigned)OUT_F);
    }

    const unsigned short* As = &lds[cur][0];
    const unsigned short* Bs = &lds[cur][TILE_USHORTS];

#pragma unroll
    for (int kk = 0; kk < 2; ++kk) {  // two 16x16x32 K-steps per BK=64 tile
      // A 16x32 frag: lanes 0-15 hold K=[0,8)+[16,24), lanes 16-31 the rest.
      Frag16 a[4];
#pragma unroll
      for (int tm = 0; tm < 4; ++tm) {
        const unsigned short* p =
            As + (wm * 64 + tm * 16 + l16) * LDS_STRIDE + kk * 32 + lhi * 8;
        a[tm].q[0] = *(const uint4*)(p);
        a[tm].q[1] = *(const uint4*)(p + 16);
      }
      // B 32x16 frag: lanes 0-15 hold K=[0,8)+[16,24), lanes 16-31 the rest, of col N.
      Frag16 b[2];
#pragma unroll
      for (int tn = 0; tn < 2; ++tn) {
        const unsigned short* p =
            Bs + (wn * 32 + tn * 16 + l16) * LDS_STRIDE + kk * 32 + lhi * 8;
        b[tn].q[0] = *(const uint4*)(p);
        b[tn].q[1] = *(const uint4*)(p + 16);
      }
#pragma unroll
      for (int tm = 0; tm < 4; ++tm)
#pragma unroll
        for (int tn = 0; tn < 2; ++tn)
          acc[tm][tn] = __builtin_amdgcn_wmma_f32_16x16x32_bf16(
              false, a[tm].v, false, b[tn].v, (short)0, acc[tm][tn], false,
              false);
    }

    // Drain this wave's outstanding TDM ops, then block-wide handshake.
    if (wave == 0) __builtin_amdgcn_s_wait_tensorcnt(0);
    __syncthreads();
  }

  // Epilogue: bias add + fp32 store (C layout: VGPR r -> M=r+8*lhi, N=l16).
  float bvals[2];
#pragma unroll
  for (int tn = 0; tn < 2; ++tn)
    bvals[tn] = Bias[bn + wn * 32 + tn * 16 + l16];

#pragma unroll
  for (int tm = 0; tm < 4; ++tm)
#pragma unroll
    for (int tn = 0; tn < 2; ++tn) {
      const int col = bn + wn * 32 + tn * 16 + l16;
#pragma unroll
      for (int r = 0; r < 8; ++r) {
        const int row = bm + wm * 64 + tm * 16 + lhi * 8 + r;
        O[(size_t)row * OUT_F + col] = acc[tm][tn][r] + bvals[tn];
      }
    }
}

// ---------------------------------------------------------------------------
// Fallback (self-contained, no workspace): round-1 on-the-fly bf16 WMMA GEMM.
// ---------------------------------------------------------------------------
#define FA_STRIDE 40
#define FB_STRIDE 72

__global__ __launch_bounds__(256) void binlin_wmma_fallback(
    const float* __restrict__ X, const float* __restrict__ W,
    const float* __restrict__ Bias, float* __restrict__ O) {
  __shared__ __align__(16) unsigned short ldsA[BM * FA_STRIDE];
  __shared__ __align__(16) unsigned short ldsB[BN * FB_STRIDE];

  const int t    = threadIdx.x;
  const int lane = t & 31;
  const int wave = t >> 5;
  const int wm   = wave >> 2;
  const int wn   = wave & 3;
  const int l16  = lane & 15;
  const int lhi  = lane >> 4;
  const int bm   = blockIdx.y * BM;
  const int bn   = blockIdx.x * BN;

  v8f acc[4][2];
#pragma unroll
  for (int tm = 0; tm < 4; ++tm)
#pragma unroll
    for (int tn = 0; tn < 2; ++tn)
      acc[tm][tn] = (v8f){};

  for (int k0 = 0; k0 < IN_F; k0 += 32) {
    __syncthreads();
#pragma unroll
    for (int i = 0; i < 8; ++i) {
      const int idx = t + i * 256;
      const int m   = idx >> 4;
      const int k   = (idx & 15) << 1;
      const float2 v = *(const float2*)(X + (size_t)(bm + m) * IN_F + (k0 + k));
      *(unsigned int*)&ldsA[m * FA_STRIDE + k] =
          sign_bf16(v.x) | (sign_bf16(v.y) << 16);
    }
#pragma unroll
    for (int i = 0; i < 8; ++i) {
      const int idx = t + i * 256;
      const int k   = idx >> 6;
      const int n   = (idx & 63) << 1;
      const float2 v = *(const float2*)(W + (size_t)(k0 + k) * OUT_F + (bn + n));
      ldsB[(n    ) * FB_STRIDE + k] = f2bf_bits(v.x);
      ldsB[(n + 1) * FB_STRIDE + k] = f2bf_bits(v.y);
    }
    __syncthreads();

    Frag16 a[4], b[2];
#pragma unroll
    for (int tm = 0; tm < 4; ++tm) {
      const unsigned short* p =
          &ldsA[(wm * 64 + tm * 16 + l16) * FA_STRIDE + lhi * 8];
      a[tm].q[0] = *(const uint4*)(p);
      a[tm].q[1] = *(const uint4*)(p + 16);
    }
#pragma unroll
    for (int tn = 0; tn < 2; ++tn) {
      const unsigned short* p =
          &ldsB[(wn * 32 + tn * 16 + l16) * FB_STRIDE + lhi * 8];
      b[tn].q[0] = *(const uint4*)(p);
      b[tn].q[1] = *(const uint4*)(p + 16);
    }
#pragma unroll
    for (int tm = 0; tm < 4; ++tm)
#pragma unroll
      for (int tn = 0; tn < 2; ++tn)
        acc[tm][tn] = __builtin_amdgcn_wmma_f32_16x16x32_bf16(
            false, a[tm].v, false, b[tn].v, (short)0, acc[tm][tn], false,
            false);
  }

  float bvals[2];
#pragma unroll
  for (int tn = 0; tn < 2; ++tn)
    bvals[tn] = Bias[bn + wn * 32 + tn * 16 + l16];
#pragma unroll
  for (int tm = 0; tm < 4; ++tm)
#pragma unroll
    for (int tn = 0; tn < 2; ++tn) {
      const int col = bn + wn * 32 + tn * 16 + l16;
#pragma unroll
      for (int r = 0; r < 8; ++r) {
        const int row = bm + wm * 64 + tm * 16 + lhi * 8 + r;
        O[(size_t)row * OUT_F + col] = acc[tm][tn][r] + bvals[tn];
      }
    }
}

extern "C" void kernel_launch(void* const* d_in, const int* in_sizes, int n_in,
                              void* d_out, int out_size, void* d_ws,
                              size_t ws_size, hipStream_t stream) {
  (void)in_sizes; (void)n_in; (void)out_size;
  const float* x    = (const float*)d_in[0];
  const float* w    = (const float*)d_in[1];
  const float* bias = (const float*)d_in[2];
  float* out        = (float*)d_out;

  const size_t xb_elems = (size_t)TOKENS * IN_F;   // 64 MB as bf16
  const size_t wt_elems = (size_t)IN_F * OUT_F;    // 32 MB as bf16
  const size_t need     = (xb_elems + wt_elems) * sizeof(unsigned short);

  const dim3 grid(OUT_F / BN, TOKENS / BM);  // 32 x 64

  if (ws_size >= need) {
    unsigned short* xb = (unsigned short*)d_ws;
    unsigned short* wt = xb + xb_elems;
    xbin_kernel<<<(int)(xb_elems / (256 * 8)), 256, 0, stream>>>(x, xb);
    wtrans_kernel<<<dim3(OUT_F / 32, IN_F / 32), 256, 0, stream>>>(w, wt);
    binlin_tdm_wmma_kernel<<<grid, dim3(256), 0, stream>>>(xb, wt, bias, out);
  } else {
    binlin_wmma_fallback<<<grid, dim3(256), 0, stream>>>(x, w, bias, out);
  }
}
